// CachedGQMultiheadAttention_88682484728633
// MI455X (gfx1250) — matrix operand order repaired and checked by reading, not compiled, not run on hardware
//
#include <hip/hip_runtime.h>
#include <hip/hip_bf16.h>

// ---------------------------------------------------------------------------
// CDNA5 (gfx1250) GQA attention block.
// f16 WMMA (f32 accumulate) for all matrix math; TDM / async-LDS / prefetch
// for data movement where the toolchain exposes them.
// ---------------------------------------------------------------------------

typedef __attribute__((ext_vector_type(16))) _Float16 v16h;
typedef __attribute__((ext_vector_type(8)))  _Float16 v8h;
typedef __attribute__((ext_vector_type(8)))  float    v8f;

typedef unsigned int u32x4 __attribute__((ext_vector_type(4)));
typedef int          i32x4 __attribute__((ext_vector_type(4)));
typedef int          i32x8 __attribute__((ext_vector_type(8)));

typedef __attribute__((address_space(3))) void LDSV;

union Frag16 { v16h v; v8h h[2]; };

#define WMMA_F16(A, Bm, C) \
  __builtin_amdgcn_wmma_f32_16x16x32_f16(false, (A), false, (Bm), (short)0, (C), false, false)

// ---- CDNA5 data-mover feature detection (compile-safe fallbacks) ----------
#if defined(__HIP_DEVICE_COMPILE__) && \
    __has_builtin(__builtin_amdgcn_tensor_load_to_lds) && \
    __has_builtin(__builtin_amdgcn_s_wait_tensorcnt)
#define HAVE_TDM 1
#else
#define HAVE_TDM 0
#endif

#if defined(__HIP_DEVICE_COMPILE__) && \
    __has_builtin(__builtin_amdgcn_global_load_async_to_lds_b128) && \
    __has_builtin(__builtin_amdgcn_s_wait_asynccnt)
#define HAVE_ASYNC 1
#else
#define HAVE_ASYNC 0
#endif

// Copy 16 bytes global -> LDS (async engine when available).
// Builtin signature (probed via diagnostics): (int4* gsrc, int4* ldst, Ii, Ii).
__device__ __forceinline__ void copy16_to_lds(const _Float16* gsrc, _Float16* ldst) {
#if HAVE_ASYNC
  __builtin_amdgcn_global_load_async_to_lds_b128((i32x4*)gsrc, (i32x4*)ldst, 0, 0);
#else
  *(v8h*)ldst = *(const v8h*)gsrc;
#endif
}

__device__ __forceinline__ void wait_async_lds() {
#if HAVE_ASYNC
  __builtin_amdgcn_s_wait_asynccnt(0);
#endif
}

#if HAVE_TDM
// TDM 2D tile load: `rows` rows of `width` f16 elements, global row stride
// `strideElems` (f16 units), packed contiguously into LDS at `ldsDst`.
// Descriptor layout per CDNA5 ISA.
__device__ __forceinline__ void tdm_load_2d(const void* gsrc, void* ldsDst,
                                            unsigned width, unsigned rows,
                                            unsigned long long strideElems) {
  unsigned long long ga = (unsigned long long)(uintptr_t)gsrc;
  unsigned lds = (unsigned)(uintptr_t)(LDSV*)ldsDst;
  u32x4 g0;
  g0[0] = 1u;                                            // count=1 (valid, user mode)
  g0[1] = lds;                                           // lds_addr (bytes)
  g0[2] = (unsigned)ga;                                  // global_addr[31:0]
  g0[3] = (unsigned)((ga >> 32) & 0x1FFFFFFull) | (2u << 30);  // ga[56:32] | type=2
  // group1: data_size=1 (2 bytes); tensor_dim0=width; tensor_dim1=rows;
  // tile_dim0=width; tile_dim1=rows; tensor_dim0_stride=strideElems.
  unsigned long long q0 = 0x10000ull |
                          ((unsigned long long)(width & 0xFFFFu) << 48);
  unsigned long long q1 = (unsigned long long)(width >> 16) |
                          ((unsigned long long)rows << 16) |
                          ((unsigned long long)(width & 0xFFFFu) << 48);
  unsigned long long q2 = (unsigned long long)(rows & 0xFFFFu) |
                          ((strideElems & 0xFFFFFFFFull) << 32);
  unsigned long long q3 = (strideElems >> 32) & 0xFFFFull;  // dim1_stride unused (2D)
  i32x8 g1;
  g1[0] = (int)q0; g1[1] = (int)(q0 >> 32);
  g1[2] = (int)q1; g1[3] = (int)(q1 >> 32);
  g1[4] = (int)q2; g1[5] = (int)(q2 >> 32);
  g1[6] = (int)q3; g1[7] = (int)(q3 >> 32);
  i32x4 z4 = {0, 0, 0, 0};
#if __clang_major__ >= 23
  i32x8 z8 = {0, 0, 0, 0, 0, 0, 0, 0};
  __builtin_amdgcn_tensor_load_to_lds(g0, g1, z4, z4, z8, 0);
#else
  __builtin_amdgcn_tensor_load_to_lds(g0, g1, z4, z4, 0);
#endif
}
#endif  // HAVE_TDM

__device__ __forceinline__ void wait_tdm() {
#if HAVE_TDM
  __builtin_amdgcn_s_wait_tensorcnt(0);
#endif
}

// Problem constants (from the reference).
constexpr int NB    = 4;
constexpr int NLQ   = 1024;
constexpr int ND    = 2048;
constexpr int NH    = 16;
constexpr int NG    = 4;
constexpr int NHD   = 128;
constexpr int NGS   = NH / NG;
constexpr int NPAST = 1024;
constexpr int NLK   = NPAST + NLQ;  // 2048
constexpr int NDK   = NG * NHD;     // 512
constexpr int NM    = NB * NLQ;     // 4096
constexpr float QSCALE = 0.08838834764831845f; // 1/sqrt(128)

// ---------------------------------------------------------------------------
// Elementwise helpers
// ---------------------------------------------------------------------------
__global__ void k_f32_to_f16(const float* __restrict__ in, _Float16* __restrict__ out, int n) {
  int i = blockIdx.x * blockDim.x + threadIdx.x;
  if (i < n) out[i] = (_Float16)in[i];
}

__global__ void k_rope_q(const float* __restrict__ qf, const float* __restrict__ rf,
                         _Float16* __restrict__ Qh) {
  int idx = blockIdx.x * blockDim.x + threadIdx.x;
  if (idx >= NB * NLQ * NH * (NHD / 2)) return;
  int i = idx & 63; int t = idx >> 6;
  int h = t % NH; t /= NH;
  int lq = t % NLQ; int b = t / NLQ;
  float f = rf[lq * (NHD / 2) + i];
  float c = __cosf(f), s = __sinf(f);
  const float* p = qf + (((size_t)b * NLQ + lq) * NH + h) * NHD + 2 * i;
  float x1 = p[0], x2 = p[1];
  _Float16* o = Qh + (((size_t)b * NH + h) * NLQ + lq) * NHD + 2 * i;
  o[0] = (_Float16)((x1 * c - x2 * s) * QSCALE);
  o[1] = (_Float16)((x1 * s + x2 * c) * QSCALE);
}

__global__ void k_rope_k(const float* __restrict__ kf, const float* __restrict__ rf,
                         _Float16* __restrict__ Kh) {
  int idx = blockIdx.x * blockDim.x + threadIdx.x;
  if (idx >= NB * NLQ * NG * (NHD / 2)) return;
  int i = idx & 63; int t = idx >> 6;
  int g = t % NG; t /= NG;
  int lq = t % NLQ; int b = t / NLQ;
  float f = rf[lq * (NHD / 2) + i];
  float c = __cosf(f), s = __sinf(f);
  const float* p = kf + (((size_t)b * NLQ + lq) * NG + g) * NHD + 2 * i;
  float x1 = p[0], x2 = p[1];
  _Float16* o = Kh + (((size_t)b * NG + g) * NLK + NPAST + lq) * NHD + 2 * i;
  o[0] = (_Float16)(x1 * c - x2 * s);
  o[1] = (_Float16)(x1 * s + x2 * c);
}

__global__ void k_pack_v(const float* __restrict__ vf, _Float16* __restrict__ Vh) {
  int idx = blockIdx.x * blockDim.x + threadIdx.x;
  if (idx >= NB * NLQ * NG * NHD) return;
  int hd = idx & (NHD - 1); int t = idx >> 7;
  int g = t % NG; t /= NG;
  int lq = t % NLQ; int b = t / NLQ;
  Vh[(((size_t)b * NG + g) * NLK + NPAST + lq) * NHD + hd] = (_Float16)vf[idx];
}

__global__ void k_pack_past(const float* __restrict__ past, _Float16* __restrict__ cache) {
  int idx = blockIdx.x * blockDim.x + threadIdx.x;
  if (idx >= NB * NG * NPAST * NHD) return;
  int hd = idx & (NHD - 1); int t = idx >> 7;
  int p = t % NPAST; t /= NPAST;
  int g = t % NG; int b = t / NG;
  cache[(((size_t)b * NG + g) * NLK + p) * NHD + hd] = (_Float16)past[idx];
}

// ---------------------------------------------------------------------------
// GEMM: C[M,N] = A[M,K] (f16) @ Bw[K,N] (f16) + bias (f32), f32 out.
// Workgroup tile 128x64, BK=32; 8 waves, each 32x32 via 2x2 WMMA frags.
// A tile staged by TDM (or async-LDS); B tile manually transposed in LDS.
// ---------------------------------------------------------------------------
template <int N>
__global__ __launch_bounds__(256) void k_gemm_bias(
    const _Float16* __restrict__ A, const _Float16* __restrict__ Bw,
    const float* __restrict__ bias, float* __restrict__ C, int K) {
  __shared__ _Float16 As[128 * 32];   // [m][k]
  __shared__ _Float16 Bst[64 * 32];   // [n][k] (transposed)

  const int tid = threadIdx.x;
  const int lane = tid & 31, w = tid >> 5;
  const int l15 = lane & 15;
  const bool hi = lane >= 16;
  const int hi8 = hi ? 8 : 0, hi16 = hi ? 16 : 0;
  const int wm = w >> 1, wn = w & 1;
  const int m0 = blockIdx.x * 128, n0 = blockIdx.y * 64;

  v8f zero = {};
  v8f acc[2][2];
  for (int mi = 0; mi < 2; ++mi)
    for (int ni = 0; ni < 2; ++ni) acc[mi][ni] = zero;

  const int arow = tid >> 1, apart = (tid & 1) * 16;
  const int bkr = tid >> 3, bnc = (tid & 7) * 8;
  (void)arow; (void)apart;

  for (int k0 = 0; k0 < K; k0 += 32) {
    __syncthreads();
#if HAVE_TDM
    if (w == 0)
      tdm_load_2d(A + (size_t)m0 * K + k0, As, 32, 128, (unsigned long long)K);
#else
    {
      const _Float16* s = A + (size_t)(m0 + arow) * K + k0 + apart;
      copy16_to_lds(s, &As[arow * 32 + apart]);
      copy16_to_lds(s + 8, &As[arow * 32 + apart + 8]);
    }
#endif
    {
      v8h bv = *(const v8h*)(Bw + (size_t)(k0 + bkr) * N + n0 + bnc);
#pragma unroll
      for (int j = 0; j < 8; ++j) Bst[(bnc + j) * 32 + bkr] = bv[j];
    }
    if (k0 + 32 < K) {  // pull next K-step into cache while we compute
      __builtin_prefetch(A + (size_t)(m0 + (tid >> 1)) * K + k0 + 32, 0, 3);
      __builtin_prefetch(Bw + (size_t)(k0 + 32 + bkr) * N + n0 + bnc, 0, 3);
    }
    wait_tdm();
    wait_async_lds();
    __syncthreads();

    Frag16 af[2], bf[2];
#pragma unroll
    for (int mi = 0; mi < 2; ++mi) {
      int r = (wm * 32 + mi * 16 + l15) * 32;
      af[mi].h[0] = *(const v8h*)&As[r + hi8];
      af[mi].h[1] = *(const v8h*)&As[r + 16 + hi8];
    }
#pragma unroll
    for (int ni = 0; ni < 2; ++ni) {
      int r = (wn * 32 + ni * 16 + l15) * 32;
      bf[ni].h[0] = *(const v8h*)&Bst[r + hi8];
      bf[ni].h[1] = *(const v8h*)&Bst[r + 16 + hi8];
    }
#pragma unroll
    for (int mi = 0; mi < 2; ++mi)
#pragma unroll
      for (int ni = 0; ni < 2; ++ni)
        acc[mi][ni] = WMMA_F16(af[mi].v, bf[ni].v, acc[mi][ni]);
  }

#pragma unroll
  for (int mi = 0; mi < 2; ++mi)
#pragma unroll
    for (int ni = 0; ni < 2; ++ni) {
      int n = n0 + wn * 32 + ni * 16 + l15;
      float bb = bias[n];
#pragma unroll
      for (int j = 0; j < 8; ++j) {
        int m = m0 + wm * 32 + mi * 16 + hi8 + j;
        C[(size_t)m * N + n] = acc[mi][ni][j] + bb;
      }
    }
}

// ---------------------------------------------------------------------------
// Flash attention: workgroup = (b, h, 64 q-rows); 4 waves x 16 q-rows.
// Q block + K tiles staged by TDM/async; V transposed manually; S^T per wave
// keeps the softmax reduction per-lane + one shfl_xor(16).
// ---------------------------------------------------------------------------
__global__ __launch_bounds__(128) void k_attn(
    const _Float16* __restrict__ Qh,  // [B,H,LQ,HD] (pre-scaled by 1/sqrt(HD))
    const _Float16* __restrict__ Kh,  // [B,G,LK,HD]
    const _Float16* __restrict__ Vh,  // [B,G,LK,HD]
    _Float16* __restrict__ Ao) {      // [B,LQ,H,HD]
  __shared__ _Float16 Qs[64 * 128];
  __shared__ _Float16 Ks[32 * 128];
  __shared__ _Float16 Vst[128 * 32];
  __shared__ _Float16 Ps[4][16 * 32];
  __shared__ float CorrS[4][16];
  __shared__ float SumS[4][16];

  const int qblk = blockIdx.x;
  const int h = blockIdx.y, b = blockIdx.z;
  const int g = h / NGS;
  const int tid = threadIdx.x;
  const int w = tid >> 5, lane = tid & 31;
  const int l15 = lane & 15;
  const bool hi = lane >= 16;
  const int hi8 = hi ? 8 : 0, hi16 = hi ? 16 : 0;
  const int qBase = qblk * 64;

  const _Float16* qsrc = Qh + (((size_t)b * NH + h) * NLQ + qBase) * NHD;
#if HAVE_TDM
  if (w == 0) tdm_load_2d(qsrc, Qs, 128, 64, 128ull);
#else
  for (int c = tid; c < 64 * 128 / 8; c += 128)
    copy16_to_lds(qsrc + c * 8, &Qs[c * 8]);
#endif

  v8f zero = {};
  v8f outAcc[8];
#pragma unroll
  for (int nt = 0; nt < 8; ++nt) outAcc[nt] = zero;
  float runM = -3.0e38f, runS = 0.0f;

  const int maxKey = NPAST + qBase + 63;
  const _Float16* kbase = Kh + ((size_t)b * NG + g) * NLK * NHD;
  const _Float16* vbase = Vh + ((size_t)b * NG + g) * NLK * NHD;

  for (int kstart = 0; kstart < NLK; kstart += 32) {
    if (kstart > maxKey) break;  // uniform across workgroup
    __syncthreads();             // prior iteration done before restaging
#if HAVE_TDM
    if (w == 0) tdm_load_2d(kbase + (size_t)kstart * NHD, Ks, 128, 32, 128ull);
#else
    for (int c = tid; c < 32 * 128 / 8; c += 128)
      copy16_to_lds(kbase + (size_t)kstart * NHD + c * 8, &Ks[c * 8]);
#endif
    // V tile: transpose into [hd][key].
    for (int c = tid; c < 32 * 16; c += 128) {
      int key = c >> 4, off = (c & 15) * 8;
      v8h vv = *(const v8h*)(vbase + ((size_t)kstart + key) * NHD + off);
#pragma unroll
      for (int j = 0; j < 8; ++j) Vst[(off + j) * 32 + key] = vv[j];
    }
    if (kstart + 32 <= maxKey) {  // prefetch next tiles into cache
      __builtin_prefetch(kbase + ((size_t)kstart + 32) * NHD + (size_t)tid * 32, 0, 3);
      __builtin_prefetch(vbase + ((size_t)kstart + 32) * NHD + (size_t)tid * 32, 0, 3);
    }
    wait_tdm();
    wait_async_lds();
    __syncthreads();

    // S^T(16 keys x 16 q) for two 16-key subtiles; HD covered in 4 WMMAs.
    v8f sv[2];
#pragma unroll
    for (int kt = 0; kt < 2; ++kt) {
      v8f acc = zero;
#pragma unroll
      for (int dk = 0; dk < 4; ++dk) {
        Frag16 a, bq;
        int kr = (kt * 16 + l15) * 128 + dk * 32;
        a.h[0] = *(const v8h*)&Ks[kr + hi8];
        a.h[1] = *(const v8h*)&Ks[kr + 16 + hi8];
        int qr = (w * 16 + l15) * 128 + dk * 32;
        bq.h[0] = *(const v8h*)&Qs[qr + hi8];
        bq.h[1] = *(const v8h*)&Qs[qr + 16 + hi8];
        acc = WMMA_F16(a.v, bq.v, acc);
      }
      sv[kt] = acc;
    }

    // Causal mask + online softmax stats (per q column = l15).
    const int qpos = NPAST + qBase + w * 16 + l15;
    float tmax = -3.0e38f;
#pragma unroll
    for (int kt = 0; kt < 2; ++kt)
#pragma unroll
      for (int j = 0; j < 8; ++j) {
        int key = kstart + kt * 16 + hi8 + j;
        if (key > qpos) sv[kt][j] = -1.0e30f;
        tmax = fmaxf(tmax, sv[kt][j]);
      }
    tmax = fmaxf(tmax, __shfl_xor(tmax, 16, 32));
    float nm = fmaxf(runM, tmax);
    float corr = __expf(runM - nm);
    float psum = 0.0f;
#pragma unroll
    for (int kt = 0; kt < 2; ++kt)
#pragma unroll
      for (int j = 0; j < 8; ++j) {
        float p = __expf(sv[kt][j] - nm);
        sv[kt][j] = p;
        psum += p;
      }
    psum += __shfl_xor(psum, 16, 32);
    runS = runS * corr + psum;
    runM = nm;

    if (lane < 16) CorrS[w][lane] = corr;
#pragma unroll
    for (int kt = 0; kt < 2; ++kt) {
      v8h ph;
#pragma unroll
      for (int j = 0; j < 8; ++j) ph[j] = (_Float16)sv[kt][j];
      *(v8h*)&Ps[w][l15 * 32 + kt * 16 + hi8] = ph;
    }
    __syncthreads();

    // Rescale running output (q is the M dimension: q_m = hi8 + j).
    float cj[8];
#pragma unroll
    for (int j = 0; j < 8; ++j) cj[j] = CorrS[w][hi8 + j];
#pragma unroll
    for (int nt = 0; nt < 8; ++nt)
#pragma unroll
      for (int j = 0; j < 8; ++j) outAcc[nt][j] *= cj[j];

    // out(16q x 128hd) += P(16q x 32key) @ V(32key x 16hd), 8 N-tiles.
    Frag16 pa;
    pa.h[0] = *(const v8h*)&Ps[w][l15 * 32 + hi8];
    pa.h[1] = *(const v8h*)&Ps[w][l15 * 32 + 16 + hi8];
#pragma unroll
    for (int nt = 0; nt < 8; ++nt) {
      Frag16 vb;
      int hd = nt * 16 + l15;
      vb.h[0] = *(const v8h*)&Vst[hd * 32 + hi8];
      vb.h[1] = *(const v8h*)&Vst[hd * 32 + 16 + hi8];
      outAcc[nt] = WMMA_F16(pa.v, vb.v, outAcc[nt]);
    }
  }

  // Normalize and store to [B,LQ,H,HD].
  if (lane < 16) SumS[w][lane] = runS;
  __syncthreads();
  float inv[8];
#pragma unroll
  for (int j = 0; j < 8; ++j) inv[j] = 1.0f / SumS[w][hi8 + j];
#pragma unroll
  for (int nt = 0; nt < 8; ++nt) {
    int hd = nt * 16 + l15;
#pragma unroll
    for (int j = 0; j < 8; ++j) {
      int q = qBase + w * 16 + hi8 + j;
      Ao[(((size_t)b * NLQ + q) * NH + h) * NHD + hd] = (_Float16)(outAcc[nt][j] * inv[j]);
    }
  }
}

// ---------------------------------------------------------------------------
// Launch
// ---------------------------------------------------------------------------
static inline int cdiv(int a, int b) { return (a + b - 1) / b; }

extern "C" void kernel_launch(void* const* d_in, const int* in_sizes, int n_in,
                              void* d_out, int out_size, void* d_ws, size_t ws_size,
                              hipStream_t stream) {
  (void)in_sizes; (void)n_in; (void)out_size; (void)ws_size;

  const float* x      = (const float*)d_in[0];
  /* d_in[1] = mask (bool), recomputed analytically */
  const float* rf     = (const float*)d_in[2];
  const float* past_k = (const float*)d_in[3];
  const float* past_v = (const float*)d_in[4];
  const float* Wq     = (const float*)d_in[5];
  const float* bq     = (const float*)d_in[6];
  const float* Wk     = (const float*)d_in[7];
  const float* bk     = (const float*)d_in[8];
  const float* Wv     = (const float*)d_in[9];
  const float* bv     = (const float*)d_in[10];
  const float* Wo     = (const float*)d_in[11];
  const float* bo     = (const float*)d_in[12];
  float* out = (float*)d_out;

  char* p = (char*)d_ws;
  _Float16* xh  = (_Float16*)p; p += (size_t)NM * ND * 2;
  _Float16* Wqh = (_Float16*)p; p += (size_t)ND * ND * 2;
  _Float16* Wkh = (_Float16*)p; p += (size_t)ND * NDK * 2;
  _Float16* Wvh = (_Float16*)p; p += (size_t)ND * NDK * 2;
  _Float16* Woh = (_Float16*)p; p += (size_t)ND * ND * 2;
  float*    qf  = (float*)p;    p += (size_t)NM * ND * 4;
  float*    kf  = (float*)p;    p += (size_t)NM * NDK * 4;
  float*    vf  = (float*)p;    p += (size_t)NM * NDK * 4;
  _Float16* Qh  = (_Float16*)p; p += (size_t)NM * ND * 2;
  _Float16* Kh  = (_Float16*)p; p += (size_t)NB * NG * NLK * NHD * 2;
  _Float16* Vh  = (_Float16*)p; p += (size_t)NB * NG * NLK * NHD * 2;
  _Float16* Ao  = (_Float16*)p; p += (size_t)NM * ND * 2;

  k_f32_to_f16<<<cdiv(NM * ND, 256), 256, 0, stream>>>(x, xh, NM * ND);
  k_f32_to_f16<<<cdiv(ND * ND, 256), 256, 0, stream>>>(Wq, Wqh, ND * ND);
  k_f32_to_f16<<<cdiv(ND * NDK, 256), 256, 0, stream>>>(Wk, Wkh, ND * NDK);
  k_f32_to_f16<<<cdiv(ND * NDK, 256), 256, 0, stream>>>(Wv, Wvh, ND * NDK);
  k_f32_to_f16<<<cdiv(ND * ND, 256), 256, 0, stream>>>(Wo, Woh, ND * ND);

  k_gemm_bias<ND ><<<dim3(NM / 128, ND / 64),  256, 0, stream>>>(xh, Wqh, bq, qf, ND);
  k_gemm_bias<NDK><<<dim3(NM / 128, NDK / 64), 256, 0, stream>>>(xh, Wkh, bk, kf, ND);
  k_gemm_bias<NDK><<<dim3(NM / 128, NDK / 64), 256, 0, stream>>>(xh, Wvh, bv, vf, ND);

  k_rope_q<<<cdiv(NB * NLQ * NH * (NHD / 2), 256), 256, 0, stream>>>(qf, rf, Qh);
  k_rope_k<<<cdiv(NB * NLQ * NG * (NHD / 2), 256), 256, 0, stream>>>(kf, rf, Kh);
  k_pack_v<<<cdiv(NB * NLQ * NG * NHD, 256), 256, 0, stream>>>(vf, Vh);
  k_pack_past<<<cdiv(NB * NG * NPAST * NHD, 256), 256, 0, stream>>>(past_k, Kh);
  k_pack_past<<<cdiv(NB * NG * NPAST * NHD, 256), 256, 0, stream>>>(past_v, Vh);

  k_attn<<<dim3(NLQ / 64, NH, NB), 128, 0, stream>>>(Qh, Kh, Vh, Ao);

  k_gemm_bias<ND><<<dim3(NM / 128, ND / 64), 256, 0, stream>>>(Ao, Woh, bo, out, ND);
}
